// GTREncoder2_6433861009965
// MI455X (gfx1250) — matrix operand order repaired and checked by reading, not compiled, not run on hardware
//
#include <hip/hip_runtime.h>
#include <hip/hip_bf16.h>

// ---------------- problem constants ----------------
constexpr int kV = 50000, kE = 256, kH = 512, kB = 32, kS = 512;
constexpr int k4H = 4 * kH;      // 2048
constexpr int k2H = 2 * kH;      // 1024
constexpr int kMrows = kS * kB;  // 16384 time-major rows

typedef __attribute__((ext_vector_type(16))) __bf16 v16bf;
typedef __attribute__((ext_vector_type(8)))  float  v8f;

union FragAB { v16bf v; uint4 q[2]; };

__device__ inline unsigned short f2bf(float f) {
  unsigned u = __float_as_uint(f);
  unsigned r = u + 0x7FFFu + ((u >> 16) & 1u);
  return (unsigned short)(r >> 16);
}
__device__ inline uint4 ldg16(const unsigned short* p) {
  return *reinterpret_cast<const uint4*>(p);
}
__device__ inline v8f wmma_bf16(v16bf a, v16bf b, v8f c) {
  return __builtin_amdgcn_wmma_f32_16x16x32_bf16(false, a, false, b, (short)0, c, false, false);
}
__device__ inline float sigm(float x) { return 1.0f / (1.0f + __expf(-x)); }
__device__ inline float tanh_fast(float x) {
  x = fminf(fmaxf(x, -15.0f), 15.0f);
  float e = __expf(2.0f * x);
  return (e - 1.0f) / (e + 1.0f);
}

// ---------------- f32 -> bf16 convert ----------------
__global__ void k_f32_to_bf16(const float* __restrict__ s, unsigned short* __restrict__ d, int n) {
  int i = blockIdx.x * 256 + threadIdx.x;
  if (i < n) d[i] = f2bf(s[i]);
}

// ---------------- embedding gather (time-major, bf16) ----------------
__global__ void k_gather(const int* __restrict__ seqs, const float* __restrict__ emb,
                         unsigned short* __restrict__ out) {
  int row = blockIdx.x;          // row = t*32 + b
  int t = row >> 5, b = row & 31;
  int e = threadIdx.x;           // 256 threads == kE
  int idx = seqs[b * kS + t];
  out[(size_t)row * kE + e] = f2bf(emb[(size_t)idx * kE + e]);
}

// ---------------- input projection GEMM: X = emb @ Wih^T + bias ----------------
// A: emb_bf [16384 x 256] (bw samples time-reversed rows), B: Wih [2048 x 256]
__global__ void k_input_proj(const unsigned short* __restrict__ embbf,
                             const unsigned short* __restrict__ wih_fw,
                             const unsigned short* __restrict__ wih_bw,
                             const float* __restrict__ bih_fw, const float* __restrict__ bhh_fw,
                             const float* __restrict__ bih_bw, const float* __restrict__ bhh_bw,
                             float* __restrict__ X) {
  const int lane = threadIdx.x & 31, lo = lane & 15, hi = lane >> 4;
  const int tile = blockIdx.x * 8 + (threadIdx.x >> 5);
  constexpr int tilesPerDir = (kMrows / 16) * (k4H / 16);  // 1024*128
  const int dir = tile / tilesPerDir;
  const int r0 = tile % tilesPerDir;
  const int m0 = (r0 / (k4H / 16)) * 16;
  const int n0 = (r0 % (k4H / 16)) * 16;

  const unsigned short* wih = dir ? wih_bw : wih_fw;
  const float* b1 = dir ? bih_bw : bih_fw;
  const float* b2 = dir ? bhh_bw : bhh_fw;

  const int n = n0 + lo;
  float bias = b1[n] + b2[n];
  v8f acc;
#pragma unroll
  for (int r = 0; r < 8; ++r) acc[r] = bias;

  int mA = m0 + lo;
  if (dir) { int t = mA >> 5, b = mA & 31; mA = ((kS - 1 - t) << 5) | b; }
  const unsigned short* arow = embbf + (size_t)mA * kE;
  const unsigned short* brow = wih + (size_t)n * kE;

#pragma unroll
  for (int kk = 0; kk < kE / 32; ++kk) {
    int k0 = kk * 32;
    FragAB A, Bf;
    int ka = k0 + hi * 8;
    A.q[0] = ldg16(arow + ka);
    A.q[1] = ldg16(arow + ka + 16);
    int kb = k0 + hi * 8;
    Bf.q[0] = ldg16(brow + kb);
    Bf.q[1] = ldg16(brow + kb + 16);
    acc = wmma_bf16(A.v, Bf.v, acc);
  }
  float* Xd = X + (size_t)dir * kMrows * k4H;
#pragma unroll
  for (int r = 0; r < 8; ++r)
    Xd[(size_t)(m0 + hi * 8 + r) * k4H + n] = acc[r];
}

// ---------------- persistent bidirectional jump-LSTM ----------------
// 16 blocks x 128 threads: blocks 0..7 forward, 8..15 backward. Each wave owns
// one 16-column h tile for all 512 steps; Whh slice resident in LDS (256KB).
__global__ __launch_bounds__(128, 1)
void k_lstm(const unsigned short* __restrict__ whh_bf,   // [2][2048][512] bf16
            const float* __restrict__ X,                 // [2][16384][2048] f32
            const int* __restrict__ seqs,                // [B][S]
            const float* __restrict__ jump,              // [B][S]
            unsigned short* __restrict__ hbuf,           // [2][2][32][512] bf16 ping-pong
            float* __restrict__ out,                     // outputs [B][S][1024] f32
            unsigned short* __restrict__ outbf,          // bf16 mirror for feature GEMM
            float* __restrict__ hmax_out,                // [2][32][512]
            float* __restrict__ cmax_out,                // [2][32][512]
            unsigned* __restrict__ cnt) {
  __shared__ unsigned short ldsW[4 * 4 * 16 * 512];  // 256 KB: [wave][gate][n][K]

  const int tid = threadIdx.x;
  const int wave = tid >> 5, lane = tid & 31, lo = lane & 15, hi = lane >> 4;
  const int blk = blockIdx.x, dir = blk >> 3, bd = blk & 7;
  const int n0 = (bd * 4 + wave) * 16;

  // ---- stage this block's Whh slice into LDS (row order [wave][gate][n][K]) ----
  {
    const unsigned short* whh = whh_bf + (size_t)dir * k4H * kH;
    for (int c = tid; c < 16384; c += 128) {       // 16384 uint4 chunks == 256KB
      int rowL = c >> 6, col4 = c & 63;            // 64 uint4 per 512-elem row
      int w2 = rowL >> 6, g = (rowL >> 4) & 3, nn = rowL & 15;
      int srcRow = g * kH + (bd * 4 + w2) * 16 + nn;
      reinterpret_cast<uint4*>(ldsW)[c] =
          reinterpret_cast<const uint4*>(whh + (size_t)srcRow * kH)[col4];
    }
  }
  // ---- zero parity-0 h buffers (both dirs), grid-cooperatively ----
  {
    uint4 z; z.x = z.y = z.z = z.w = 0;
    for (int c = blk * 128 + tid; c < 4096; c += 2048) {  // 2*32*512 bf16 / 8
      int d2 = c / 2048, off = c % 2048;
      reinterpret_cast<uint4*>(hbuf + (size_t)d2 * 2 * kB * kH)[off] = z;
    }
  }
  __syncthreads();

  unsigned bar_target = 16;
  // grid barrier helper (inlined): release, count, spin, acquire
  auto grid_barrier = [&](unsigned tgt) {
    __threadfence();
    __syncthreads();
    if (tid == 0) {
      atomicAdd(cnt, 1u);
      while (__hip_atomic_load(cnt, __ATOMIC_RELAXED, __HIP_MEMORY_SCOPE_AGENT) < tgt)
        __builtin_amdgcn_s_sleep(1);
      __threadfence();
    }
    __syncthreads();
  };
  grid_barrier(bar_target);  // h0/c0 visible everywhere

  const float* Xd = X + (size_t)dir * kMrows * k4H;
  float c_reg[16], hmx[16], cmx[16];
#pragma unroll
  for (int i = 0; i < 16; ++i) { c_reg[i] = 0.0f; hmx[i] = -3.4e38f; cmx[i] = -3.4e38f; }

  for (int t = 0; t < kS; ++t) {
    const int p = t & 1;
    const unsigned short* hp = hbuf + ((size_t)dir * 2 + p) * kB * kH;
    unsigned short* hn = hbuf + ((size_t)dir * 2 + (p ^ 1)) * kB * kH;
    const float* Xt = Xd + (size_t)t * kB * k4H;

    v8f acc[2][4];
#pragma unroll
    for (int mt = 0; mt < 2; ++mt)
#pragma unroll
      for (int g = 0; g < 4; ++g)
#pragma unroll
        for (int r = 0; r < 8; ++r)
          acc[mt][g][r] = Xt[(size_t)(mt * 16 + hi * 8 + r) * k4H + g * kH + n0 + lo];

#pragma unroll 4
    for (int kk = 0; kk < kH / 32; ++kk) {
      int k0 = kk * 32, ka = k0 + hi * 8;
      FragAB A0, A1;
      A0.q[0] = ldg16(hp + (size_t)lo * kH + ka);
      A0.q[1] = ldg16(hp + (size_t)lo * kH + ka + 16);
      A1.q[0] = ldg16(hp + (size_t)(16 + lo) * kH + ka);
      A1.q[1] = ldg16(hp + (size_t)(16 + lo) * kH + ka + 16);
#pragma unroll
      for (int g = 0; g < 4; ++g) {
        FragAB Bf;
        const unsigned short* bp = &ldsW[(((wave * 4 + g) * 16 + lo) * kH) + k0 + hi * 8];
        Bf.q[0] = *reinterpret_cast<const uint4*>(bp);
        Bf.q[1] = *reinterpret_cast<const uint4*>(bp + 16);
        acc[0][g] = wmma_bf16(A0.v, Bf.v, acc[0][g]);
        acc[1][g] = wmma_bf16(A1.v, Bf.v, acc[1][g]);
      }
    }

    // ---- LSTM cell elementwise + jump gate + masked output write ----
    const int s = dir ? (kS - 1 - t) : t;
#pragma unroll
    for (int mt = 0; mt < 2; ++mt)
#pragma unroll
      for (int r = 0; r < 8; ++r) {
        const int idx = mt * 8 + r;
        const int m = mt * 16 + hi * 8 + r;  // batch index
        float ig = sigm(acc[mt][0][r]);
        float fg = sigm(acc[mt][1][r]);
        float gg = tanh_fast(acc[mt][2][r]);
        float og = sigm(acc[mt][3][r]);
        float cn = fg * c_reg[idx] + ig * gg;
        float hv = og * tanh_fast(cn);
        hmx[idx] = fmaxf(hmx[idx], hv);
        cmx[idx] = fmaxf(cmx[idx], cn);
        float j = jump[(size_t)m * kS + s];
        float msk = (seqs[(size_t)m * kS + s] > 0) ? 1.0f : 0.0f;
        size_t oidx = ((size_t)m * kS + s) * k2H + dir * kH + n0 + lo;
        float ov = hv * msk;
        out[oidx] = ov;
        outbf[oidx] = f2bf(ov);
        c_reg[idx] = cn * j;
        hn[(size_t)m * kH + n0 + lo] = f2bf(hv * j);
      }

    bar_target += 16;
    grid_barrier(bar_target);
  }

#pragma unroll
  for (int mt = 0; mt < 2; ++mt)
#pragma unroll
    for (int r = 0; r < 8; ++r) {
      const int m = mt * 16 + hi * 8 + r;
      size_t o = ((size_t)dir * kB + m) * kH + n0 + lo;
      hmax_out[o] = hmx[mt * 8 + r];
      cmax_out[o] = cmx[mt * 8 + r];
    }
}

// ---------------- encoder_feature GEMM: feat = outputs @ W_h^T ----------------
__global__ void k_feature(const unsigned short* __restrict__ outbf,   // [16384][1024] bf16
                          const unsigned short* __restrict__ whbf,    // [1024][1024] bf16
                          float* __restrict__ feat) {
  const int lane = threadIdx.x & 31, lo = lane & 15, hi = lane >> 4;
  const int tile = blockIdx.x * 8 + (threadIdx.x >> 5);
  const int m0 = (tile / (k2H / 16)) * 16;
  const int n0 = (tile % (k2H / 16)) * 16;

  v8f acc = {};
  const unsigned short* arow = outbf + (size_t)(m0 + lo) * k2H;
  const unsigned short* brow = whbf + (size_t)(n0 + lo) * k2H;
#pragma unroll 8
  for (int kk = 0; kk < k2H / 32; ++kk) {
    int k0 = kk * 32;
    FragAB A, Bf;
    int ka = k0 + hi * 8;
    A.q[0] = ldg16(arow + ka);
    A.q[1] = ldg16(arow + ka + 16);
    int kb = k0 + hi * 8;
    Bf.q[0] = ldg16(brow + kb);
    Bf.q[1] = ldg16(brow + kb + 16);
    acc = wmma_bf16(A.v, Bf.v, acc);
  }
#pragma unroll
  for (int r = 0; r < 8; ++r)
    feat[(size_t)(m0 + hi * 8 + r) * k2H + n0 + lo] = acc[r];
}

// ---------------- host ----------------
extern "C" void kernel_launch(void* const* d_in, const int* in_sizes, int n_in,
                              void* d_out, int out_size, void* d_ws, size_t ws_size,
                              hipStream_t stream) {
  (void)in_sizes; (void)n_in; (void)out_size; (void)ws_size;
  const int*   seqs    = (const int*)  d_in[0];
  // d_in[1] = src_lengths (unused by reference math)
  const float* jump    = (const float*)d_in[2];
  const float* emb     = (const float*)d_in[3];
  const float* Wih_fw  = (const float*)d_in[4];
  const float* Whh_fw  = (const float*)d_in[5];
  const float* bih_fw  = (const float*)d_in[6];
  const float* bhh_fw  = (const float*)d_in[7];
  const float* Wih_bw  = (const float*)d_in[8];
  const float* Whh_bw  = (const float*)d_in[9];
  const float* bih_bw  = (const float*)d_in[10];
  const float* bhh_bw  = (const float*)d_in[11];
  const float* W_h     = (const float*)d_in[12];

  // output regions
  float* out_outputs = (float*)d_out;                       // 16384*1024
  float* out_feature = out_outputs + (size_t)kMrows * k2H;  // 16384*1024
  float* out_hmax    = out_feature + (size_t)kMrows * k2H;  // 2*32*512
  float* out_cmax    = out_hmax + (size_t)2 * kB * kH;

  // workspace layout
  char* ws = (char*)d_ws;
  size_t off = 0;
  auto alloc = [&](size_t bytes) { char* p = ws + off; off += (bytes + 255) & ~(size_t)255; return p; };
  unsigned short* emb_bf  = (unsigned short*)alloc((size_t)kMrows * kE * 2);      // 8 MB
  unsigned short* wihfw_bf= (unsigned short*)alloc((size_t)k4H * kE * 2);
  unsigned short* wihbw_bf= (unsigned short*)alloc((size_t)k4H * kE * 2);
  unsigned short* whh_bf  = (unsigned short*)alloc((size_t)2 * k4H * kH * 2);     // fw then bw
  unsigned short* wh_bf   = (unsigned short*)alloc((size_t)k2H * k2H * 2);
  unsigned*       cnt     = (unsigned*)alloc(256);
  unsigned short* hbuf    = (unsigned short*)alloc((size_t)2 * 2 * kB * kH * 2);  // ping-pong
  unsigned short* out_bf  = (unsigned short*)alloc((size_t)kMrows * k2H * 2);     // 33.5 MB
  float*          X       = (float*)alloc((size_t)2 * kMrows * k4H * 4);          // 268 MB

  hipMemsetAsync(cnt, 0, 256, stream);

  // weight conversions to bf16
  auto cvt = [&](const float* s, unsigned short* d, int n) {
    k_f32_to_bf16<<<(n + 255) / 256, 256, 0, stream>>>(s, d, n);
  };
  cvt(Wih_fw, wihfw_bf, k4H * kE);
  cvt(Wih_bw, wihbw_bf, k4H * kE);
  cvt(Whh_fw, whh_bf,              k4H * kH);
  cvt(Whh_bw, whh_bf + k4H * kH,   k4H * kH);
  cvt(W_h,    wh_bf,   k2H * k2H);

  // embedding gather (time-major bf16)
  k_gather<<<kMrows, kE, 0, stream>>>(seqs, emb, emb_bf);

  // input projections, both directions: 2*1024*128 tiles / 8 waves per block
  k_input_proj<<<(2 * (kMrows / 16) * (k4H / 16)) / 8, 256, 0, stream>>>(
      emb_bf, wihfw_bf, wihbw_bf, bih_fw, bhh_fw, bih_bw, bhh_bw, X);

  // persistent bidirectional LSTM: 16 blocks (8/dir) x 128 threads
  k_lstm<<<16, 128, 0, stream>>>(whh_bf, X, seqs, jump, hbuf,
                                 out_outputs, out_bf, out_hmax, out_cmax, cnt);

  // encoder_feature GEMM: 1024*64 tiles / 8 waves per block
  k_feature<<<((kMrows / 16) * (k2H / 16)) / 8, 256, 0, stream>>>(out_bf, wh_bf, out_feature);
}
